// Encoder_68753836474771
// MI455X (gfx1250) — matrix operand order repaired and checked by reading, not compiled, not run on hardware
//
#include <hip/hip_runtime.h>
#include <hip/hip_bf16.h>
#include <stdint.h>

// ---------------------------------------------------------------------------
// Tripletformer encoder for MI455X (gfx1250, wave32, WMMA).
// All dense contractions run through v_wmma_f32_16x16x32_bf16 (f32 accum).
// Problem constants from setup_inputs():
#define B_   16
#define L_   512
#define D_   41
#define LD_  (L_*D_)
#define N_   4096          // obs_len
#define DK_  128
#define NL_  3
#define NH_  4
#define DH_  32
#define DKDK (DK_*DK_)

typedef __attribute__((ext_vector_type(16))) __bf16 v16bf;
typedef __attribute__((ext_vector_type(8)))  float  v8f;

union FragBF { v16bf v; unsigned int u[8]; };

__device__ __forceinline__ unsigned short f2bf(float x){
  unsigned int u = __float_as_uint(x);
  u += 0x7fffu + ((u >> 16) & 1u);            // round-to-nearest-even
  return (unsigned short)(u >> 16);
}
__device__ __forceinline__ float bf2f(unsigned short s){
  return __uint_as_float(((unsigned int)s) << 16);
}
__device__ __forceinline__ v8f v8f_zero(){
  v8f z = {0.f,0.f,0.f,0.f,0.f,0.f,0.f,0.f};
  return z;
}

// A-matrix 16x32 bf16 fragment (ISA 7.12.2): lane%16 = M row, lane/16 = h.
// VGPR v holds K pair 2v+16h (contiguous, matches B fragment).
// `base` points at element [row][k0] of a row-major bf16 matrix.
__device__ __forceinline__ void load_afrag(FragBF& f, const unsigned short* base, int h){
  f.u[0] = *(const unsigned int*)(base +  0 + 16*h);
  f.u[1] = *(const unsigned int*)(base +  2 + 16*h);
  f.u[2] = *(const unsigned int*)(base +  4 + 16*h);
  f.u[3] = *(const unsigned int*)(base +  6 + 16*h);
  f.u[4] = *(const unsigned int*)(base +  8 + 16*h);
  f.u[5] = *(const unsigned int*)(base + 10 + 16*h);
  f.u[6] = *(const unsigned int*)(base + 12 + 16*h);
  f.u[7] = *(const unsigned int*)(base + 14 + 16*h);
}

// B-matrix 32x16 bf16 fragment: lane%16 = N col, lane/16 = h, VGPR v holds
// K pair 2v+16h. Source staged [n][k] (k-contiguous) so each VGPR is one
// aligned 32-bit read.  `base` = &src[n*ld + k0 + 16*h].
__device__ __forceinline__ void load_bfrag(FragBF& f, const unsigned short* base){
#pragma unroll
  for (int v = 0; v < 8; ++v) f.u[v] = *(const unsigned int*)(base + 2*v);
}

__device__ __forceinline__ v8f wmma_bf(const FragBF& a, const FragBF& b, v8f c){
  return __builtin_amdgcn_wmma_f32_16x16x32_bf16(false, a.v, false, b.v,
                                                 (short)0, c, false, false);
}

// ---------------------------------------------------------------------------
// Stable "ones first, then zeros" compaction of mask over L*D slots -> N_.
__global__ __launch_bounds__(256) void k_compact(
    const float* __restrict__ ctx, const float* __restrict__ val,
    const float* __restrict__ mask,
    float* __restrict__ tt, float* __restrict__ uu,
    float* __restrict__ mkb, int* __restrict__ cc)
{
  __shared__ int sscan[256];
  __shared__ int sbase;
  const int b = blockIdx.x, tid = threadIdx.x;
  if (tid == 0) sbase = 0;
  __syncthreads();
  for (int pass = 0; pass < 2; ++pass){
    for (int c0 = 0; c0 < LD_; c0 += 256){
      int j = c0 + tid;
      int f = 0; float mv = 0.f;
      if (j < LD_){
        mv = mask[(size_t)b*LD_ + j];
        int one = (mv > 0.f);
        f = (pass == 0) ? one : !one;
      }
      sscan[tid] = f;
      __syncthreads();
      for (int o = 1; o < 256; o <<= 1){
        int t = (tid >= o) ? sscan[tid - o] : 0;
        __syncthreads();
        sscan[tid] += t;
        __syncthreads();
      }
      int pos = sbase + sscan[tid] - f;
      if (f && pos < N_){
        size_t g = (size_t)b*N_ + pos;
        tt[g]  = ctx[(size_t)b*L_ + j / D_];
        uu[g]  = val[(size_t)b*LD_ + j];
        cc[g]  = j % D_;
        mkb[g] = (pass == 0) ? mv : 0.f;
      }
      int tot = sscan[255];
      __syncthreads();
      if (tid == 0) sbase += tot;
      __syncthreads();
    }
  }
}

// f32 -> bf16 bulk conversion (weights / inducing points).
__global__ void k_wcvt(const float* __restrict__ s, unsigned short* __restrict__ d, int n){
  int i = blockIdx.x * blockDim.x + threadIdx.x;
  if (i < n) d[i] = f2bf(s[i]);
}

// Z0 = relu(X @ iff_w + b) * mk, exploiting one-hot structure of X.
__global__ __launch_bounds__(256) void k_embed(
    const float* __restrict__ tt, const float* __restrict__ uu,
    const float* __restrict__ mkb, const int* __restrict__ cc,
    const float* __restrict__ iffw, const float* __restrict__ iffb,
    float* __restrict__ Z, unsigned short* __restrict__ Zbf,
    float* __restrict__ outmk)
{
  int row = blockIdx.x * 2 + (threadIdx.x >> 7);
  int c   = threadIdx.x & 127;
  float t = tt[row], u = uu[row], m = mkb[row];
  int ch  = cc[row];
  float v = iffw[ch*DK_ + c] + t*iffw[41*DK_ + c] + u*iffw[42*DK_ + c] + iffb[c];
  v = fmaxf(v, 0.f) * m;
  size_t idx = (size_t)row*DK_ + c;
  Z[idx] = v;
  Zbf[idx] = f2bf(v);
  if (c == 0) outmk[row] = m;
}

// ---------------------------------------------------------------------------
// Generic 128-col GEMM: per block 128 rows x 128 cols, 8 waves x (16x128).
// mode 0: out = A@W + bias
// mode 1: out = resid + relu(A@W + bias)
// mode 2: out = (resid + relu(A@W + bias)) * mk[row] + outF[idx]   (Z update)
__global__ __launch_bounds__(256) void k_gemm(
    const unsigned short* __restrict__ A, const unsigned short* __restrict__ W,
    const float* __restrict__ bias, const float* __restrict__ resid,
    const float* __restrict__ mk, float* __restrict__ outF,
    unsigned short* __restrict__ outB, int mode)
{
  __shared__ unsigned short Wt[DKDK];   // W transposed: [n][k], k-contiguous
  __shared__ float bsh[DK_];
  const int tid = threadIdx.x, w = tid >> 5, lane = tid & 31;
  const int lm = lane & 15, lh = lane >> 4;

  for (int idx = tid; idx < DKDK; idx += 256){
    int k = idx >> 7, n = idx & 127;
    Wt[n*DK_ + k] = W[idx];
  }
  if (tid < DK_) bsh[tid] = bias[tid];
  __syncthreads();

  const size_t rowbase = (size_t)blockIdx.x * 128 + 16*w;
  v8f acc[8];
#pragma unroll
  for (int nt = 0; nt < 8; ++nt) acc[nt] = v8f_zero();

#pragma unroll
  for (int kt = 0; kt < 4; ++kt){
    FragBF af;
    load_afrag(af, A + (rowbase + lm)*DK_ + kt*32, lh);
#pragma unroll
    for (int nt = 0; nt < 8; ++nt){
      FragBF bf;
      load_bfrag(bf, &Wt[(nt*16 + lm)*DK_ + kt*32 + 16*lh]);
      acc[nt] = wmma_bf(af, bf, acc[nt]);
    }
  }

#pragma unroll
  for (int nt = 0; nt < 8; ++nt){
#pragma unroll
    for (int r = 0; r < 8; ++r){
      size_t row = rowbase + r + 8*lh;
      int col = nt*16 + lm;
      size_t idx = row*DK_ + col;
      float v = acc[nt][r] + bsh[col];
      if (mode >= 1) v = resid[idx] + fmaxf(v, 0.f);
      if (mode == 2) v = v * mk[row] + outF[idx];
      if (outF) outF[idx] = v;
      if (outB) outB[idx] = f2bf(v);
    }
  }
}

// ---------------------------------------------------------------------------
// MAB0 attention: 128 inducing queries x 4096 keys, flash-style online
// softmax held in registers (row = accum element r + 8*(lane/16)).
__global__ __launch_bounds__(256) void k_attn0(
    const unsigned short* __restrict__ q0bf, const float* __restrict__ q0pre,
    const unsigned short* __restrict__ Kp, const unsigned short* __restrict__ Vp,
    const float* __restrict__ mkb,
    float* __restrict__ h0pre, unsigned short* __restrict__ h0bf)
{
  __shared__ unsigned short Ks[64*DH_];   // [key][dh]
  __shared__ unsigned short Vt[DH_*64];   // [dh][key]
  __shared__ unsigned short Ps[128*64];   // [qrow][key] bf16
  __shared__ float mkc[64];
  const int b = blockIdx.x, hh = blockIdx.y;
  const int tid = threadIdx.x, w = tid >> 5, lane = tid & 31;
  const int lm = lane & 15, lh = lane >> 4;
  const float scale = 0.0883883476483184f;   // 1/sqrt(128)

  FragBF qf;
  load_afrag(qf, q0bf + (size_t)(16*w + lm)*DK_ + hh*DH_, lh);

  float m_[8], l_[8];
  v8f O0 = v8f_zero(), O1 = v8f_zero();
#pragma unroll
  for (int r = 0; r < 8; ++r){ m_[r] = -3.0e38f; l_[r] = 0.f; }

  const size_t kvbase = (size_t)b*N_*DK_ + hh*DH_;
  for (int ch = 0; ch < N_/64; ++ch){
    const int k0 = ch*64;
    for (int idx = tid; idx < 64*DH_; idx += 256){
      int key = idx >> 5, d = idx & 31;
      size_t g = kvbase + (size_t)(k0 + key)*DK_ + d;
      Ks[key*DH_ + d] = Kp[g];
      Vt[d*64 + key]  = Vp[g];
    }
    if (tid < 64) mkc[tid] = mkb[(size_t)b*N_ + k0 + tid];
    __syncthreads();

    // S = Q K^T  (16 rows x 64 keys per wave)
    v8f sacc[4];
#pragma unroll
    for (int nt = 0; nt < 4; ++nt){
      FragBF bf;
      load_bfrag(bf, &Ks[(nt*16 + lm)*DH_ + 16*lh]);
      sacc[nt] = wmma_bf(qf, bf, v8f_zero());
    }

    // online softmax (per row r+8h, reduce across 16-lane half)
#pragma unroll
    for (int r = 0; r < 8; ++r){
      float sv[4]; float mx = m_[r];
#pragma unroll
      for (int nt = 0; nt < 4; ++nt){
        float s = sacc[nt][r];
        s = (mkc[nt*16 + lm] > 0.f) ? s*scale : -1.0e10f;
        sv[nt] = s; mx = fmaxf(mx, s);
      }
#pragma unroll
      for (int d = 1; d < 16; d <<= 1) mx = fmaxf(mx, __shfl_xor(mx, d, 32));
      float resc = __expf(m_[r] - mx);
      float lsum = 0.f;
      int row_l = 16*w + r + 8*lh;
#pragma unroll
      for (int nt = 0; nt < 4; ++nt){
        float p = __expf(sv[nt] - mx);
        lsum += p;
        Ps[row_l*64 + nt*16 + lm] = f2bf(p);
      }
#pragma unroll
      for (int d = 1; d < 16; d <<= 1) lsum += __shfl_xor(lsum, d, 32);
      l_[r] = l_[r]*resc + lsum;
      m_[r] = mx;
      O0[r] *= resc; O1[r] *= resc;
    }
    __syncthreads();

    // O += P V  (contraction over 64 keys = 2 K-steps)
#pragma unroll
    for (int kt = 0; kt < 2; ++kt){
      FragBF pa;
      load_afrag(pa, &Ps[(16*w + lm)*64 + kt*32], lh);
      FragBF v0f, v1f;
      load_bfrag(v0f, &Vt[(lm)*64      + kt*32 + 16*lh]);
      load_bfrag(v1f, &Vt[(16 + lm)*64 + kt*32 + 16*lh]);
      O0 = wmma_bf(pa, v0f, O0);
      O1 = wmma_bf(pa, v1f, O1);
    }
    __syncthreads();
  }

#pragma unroll
  for (int r = 0; r < 8; ++r){
    int row = 16*w + r + 8*lh;
    float invl = 1.f / l_[r];
    int c0 = hh*DH_ + lm, c1 = c0 + 16;
    float v0 = q0pre[row*DK_ + c0] + O0[r]*invl;
    float v1 = q0pre[row*DK_ + c1] + O1[r]*invl;
    size_t o = ((size_t)b*DK_ + row)*DK_;
    h0pre[o + c0] = v0;  h0pre[o + c1] = v1;
    h0bf [o + c0] = f2bf(v0);  h0bf[o + c1] = f2bf(v1);
  }
}

// ---------------------------------------------------------------------------
// MAB1 attention: 4096 queries x 128 keys; single-pass softmax in registers.
__global__ __launch_bounds__(256) void k_attn1(
    const unsigned short* __restrict__ q1bf, const unsigned short* __restrict__ k1bf,
    const unsigned short* __restrict__ v1bf,
    float* __restrict__ o1pre, unsigned short* __restrict__ o1bf)
{
  __shared__ unsigned short Ks[DK_*DH_];   // [key][dh]
  __shared__ unsigned short Vt[DH_*DK_];   // [dh][key]
  __shared__ unsigned short Ps[DK_*DK_];   // [qrow][key]
  const int b  = blockIdx.x >> 5;
  const int qc = blockIdx.x & 31;
  const int hh = blockIdx.y;
  const int tid = threadIdx.x, w = tid >> 5, lane = tid & 31;
  const int lm = lane & 15, lh = lane >> 4;
  const float scale = 0.0883883476483184f;
  const size_t hb = (size_t)b*DK_*DK_;

  for (int idx = tid; idx < DK_*DH_; idx += 256){
    int key = idx >> 5, d = idx & 31;
    size_t g = hb + (size_t)key*DK_ + hh*DH_ + d;
    Ks[key*DH_ + d] = k1bf[g];
    Vt[d*DK_ + key] = v1bf[g];
  }
  __syncthreads();

  const size_t q0g = (size_t)b*N_ + qc*128;
  FragBF qf;
  load_afrag(qf, q1bf + (q0g + 16*w + lm)*DK_ + hh*DH_, lh);

  v8f sacc[8];
#pragma unroll
  for (int nt = 0; nt < 8; ++nt){
    FragBF bf;
    load_bfrag(bf, &Ks[(nt*16 + lm)*DH_ + 16*lh]);
    sacc[nt] = wmma_bf(qf, bf, v8f_zero());
  }

  float linv[8];
#pragma unroll
  for (int r = 0; r < 8; ++r){
    float mx = -3.0e38f;
#pragma unroll
    for (int nt = 0; nt < 8; ++nt) mx = fmaxf(mx, sacc[nt][r]*scale);
#pragma unroll
    for (int d = 1; d < 16; d <<= 1) mx = fmaxf(mx, __shfl_xor(mx, d, 32));
    float lsum = 0.f;
    int row_l = 16*w + r + 8*lh;
#pragma unroll
    for (int nt = 0; nt < 8; ++nt){
      float p = __expf(sacc[nt][r]*scale - mx);
      lsum += p;
      Ps[row_l*DK_ + nt*16 + lm] = f2bf(p);
    }
#pragma unroll
    for (int d = 1; d < 16; d <<= 1) lsum += __shfl_xor(lsum, d, 32);
    linv[r] = 1.f / lsum;
  }
  __syncthreads();

  v8f O0 = v8f_zero(), O1 = v8f_zero();
#pragma unroll
  for (int kt = 0; kt < 4; ++kt){
    FragBF pa;
    load_afrag(pa, &Ps[(16*w + lm)*DK_ + kt*32], lh);
    FragBF v0f, v1f;
    load_bfrag(v0f, &Vt[(lm)*DK_      + kt*32 + 16*lh]);
    load_bfrag(v1f, &Vt[(16 + lm)*DK_ + kt*32 + 16*lh]);
    O0 = wmma_bf(pa, v0f, O0);
    O1 = wmma_bf(pa, v1f, O1);
  }

#pragma unroll
  for (int r = 0; r < 8; ++r){
    size_t rowg = q0g + 16*w + r + 8*lh;
    int c0 = hh*DH_ + lm, c1 = c0 + 16;
    float v0 = bf2f(q1bf[rowg*DK_ + c0]) + O0[r]*linv[r];
    float v1 = bf2f(q1bf[rowg*DK_ + c1]) + O1[r]*linv[r];
    o1pre[rowg*DK_ + c0] = v0;  o1pre[rowg*DK_ + c1] = v1;
    o1bf [rowg*DK_ + c0] = f2bf(v0);  o1bf[rowg*DK_ + c1] = f2bf(v1);
  }
}

// ---------------------------------------------------------------------------
extern "C" void kernel_launch(void* const* d_in, const int* in_sizes, int n_in,
                              void* d_out, int out_size, void* d_ws, size_t ws_size,
                              hipStream_t stream)
{
  (void)in_sizes; (void)n_in; (void)out_size; (void)ws_size;
  const float* ctx  = (const float*)d_in[0];
  const float* val  = (const float*)d_in[1];
  const float* mask = (const float*)d_in[2];
  const float* iffw = (const float*)d_in[4];
  const float* iffb = (const float*)d_in[5];
  const float* indp = (const float*)d_in[6];
  const float* Wf[8]; const float* Bf[8];
  for (int j = 0; j < 8; ++j){        // q0,k0,v0,o0,q1,k1,v1,o1
    Wf[j] = (const float*)d_in[7 + 2*j];
    Bf[j] = (const float*)d_in[8 + 2*j];
  }

  float* Z     = (float*)d_out;                       // (B,N,128)
  float* outmk = (float*)d_out + (size_t)B_*N_*DK_;   // (B,N)

  char* ws = (char*)d_ws;
  size_t off = 0;
  auto alloc = [&](size_t bytes)->void*{
    void* p = ws + off;
    off += (bytes + 255) & ~(size_t)255;
    return p;
  };
  const size_t BN = (size_t)B_*N_;
  float*          tt    = (float*)alloc(sizeof(float)*BN);
  float*          uu    = (float*)alloc(sizeof(float)*BN);
  float*          mkb   = (float*)alloc(sizeof(float)*BN);
  int*            cc    = (int*)  alloc(sizeof(int)*BN);
  unsigned short* Zbf   = (unsigned short*)alloc(2*BN*DK_);
  unsigned short* Kp    = (unsigned short*)alloc(2*BN*DK_);
  unsigned short* Vp    = (unsigned short*)alloc(2*BN*DK_);
  unsigned short* Q1bf  = (unsigned short*)alloc(2*BN*DK_);
  unsigned short* O1bf  = (unsigned short*)alloc(2*BN*DK_);
  float*          O1pre = (float*)alloc(sizeof(float)*BN*DK_);
  unsigned short* wbf   = (unsigned short*)alloc(2*(size_t)8*NL_*DKDK);
  unsigned short* indbf = (unsigned short*)alloc(2*(size_t)NL_*DKDK);
  float*          Q0pre = (float*)alloc(sizeof(float)*NL_*DKDK);
  unsigned short* Q0bf  = (unsigned short*)alloc(2*(size_t)NL_*DKDK);
  float*          H0pre = (float*)alloc(sizeof(float)*B_*DKDK);
  unsigned short* H0bf  = (unsigned short*)alloc(2*(size_t)B_*DKDK);
  unsigned short* Hbf   = (unsigned short*)alloc(2*(size_t)B_*DKDK);
  unsigned short* K1bf  = (unsigned short*)alloc(2*(size_t)B_*DKDK);
  unsigned short* V1bf  = (unsigned short*)alloc(2*(size_t)B_*DKDK);

  auto wmat = [&](int m, int i){ return wbf + ((size_t)m*NL_ + i)*DKDK; };

  // 1) stable compaction + gathers
  k_compact<<<B_, 256, 0, stream>>>(ctx, val, mask, tt, uu, mkb, cc);

  // 2) weight conversions to bf16
  const int ncvt = NL_*DKDK;
  for (int j = 0; j < 8; ++j)
    k_wcvt<<<(ncvt + 255)/256, 256, 0, stream>>>(Wf[j], wbf + (size_t)j*ncvt, ncvt);
  k_wcvt<<<(ncvt + 255)/256, 256, 0, stream>>>(indp, indbf, ncvt);

  // 3) embedding -> Z0 (f32 in d_out) + bf16 shadow; also writes mk output
  k_embed<<<B_*N_/2, 256, 0, stream>>>(tt, uu, mkb, cc, iffw, iffb, Z, Zbf, outmk);

  // 4) batch-independent MAB0 query projection: Q0 = I @ q0w + q0b
  for (int i = 0; i < NL_; ++i)
    k_gemm<<<1, 256, 0, stream>>>(indbf + (size_t)i*DKDK, wmat(0, i), Bf[0] + i*DK_,
                                  nullptr, nullptr, Q0pre + (size_t)i*DKDK,
                                  Q0bf + (size_t)i*DKDK, 0);

  const int GBIG = (int)(BN/128);   // 512 blocks
  const int GSML = B_*DK_/128;      // 16 blocks
  for (int i = 0; i < NL_; ++i){
    // MAB0: K/V projections over Z
    k_gemm<<<GBIG, 256, 0, stream>>>(Zbf, wmat(1, i), Bf[1] + i*DK_,
                                     nullptr, nullptr, nullptr, Kp, 0);
    k_gemm<<<GBIG, 256, 0, stream>>>(Zbf, wmat(2, i), Bf[2] + i*DK_,
                                     nullptr, nullptr, nullptr, Vp, 0);
    // flash attention (128 q x 4096 k), residual Q0 added
    k_attn0<<<dim3(B_, NH_), 256, 0, stream>>>(Q0bf + (size_t)i*DKDK,
                                               Q0pre + (size_t)i*DKDK,
                                               Kp, Vp, mkb, H0pre, H0bf);
    // H = H0 + relu(H0 @ o0w + o0b)
    k_gemm<<<GSML, 256, 0, stream>>>(H0bf, wmat(3, i), Bf[3] + i*DK_,
                                     H0pre, nullptr, nullptr, Hbf, 1);
    // MAB1 projections
    k_gemm<<<GBIG, 256, 0, stream>>>(Zbf, wmat(4, i), Bf[4] + i*DK_,
                                     nullptr, nullptr, nullptr, Q1bf, 0);
    k_gemm<<<GSML, 256, 0, stream>>>(Hbf, wmat(5, i), Bf[5] + i*DK_,
                                     nullptr, nullptr, nullptr, K1bf, 0);
    k_gemm<<<GSML, 256, 0, stream>>>(Hbf, wmat(6, i), Bf[6] + i*DK_,
                                     nullptr, nullptr, nullptr, V1bf, 0);
    // attention (4096 q x 128 k), residual Q1 added
    k_attn1<<<dim3(B_*(N_/128), NH_), 256, 0, stream>>>(Q1bf, K1bf, V1bf,
                                                        O1pre, O1bf);
    // Z = (O1 + relu(O1 @ o1w + o1b)) * mk + Z   (in place on d_out)
    k_gemm<<<GBIG, 256, 0, stream>>>(O1bf, wmat(7, i), Bf[7] + i*DK_,
                                     O1pre, mkb, Z, Zbf, 2);
  }
}
